// RNN_32650341384689
// MI455X (gfx1250) — compile-verified, matrix operand order checked by reading, not run on hardware
//
#include <hip/hip_runtime.h>
#include <hip/hip_bf16.h>

#define VOCAB  32000
#define EMBED  512
#define HIDDEN 1024
#define OUTD   32000
#define BATCH  64
#define TLEN   512
#define KTOT   (EMBED + HIDDEN)   // 1536
#define KSPLIT 4                  // waves per step-block, K split 4 ways
#define KCHUNK (KTOT / KSPLIT)    // 384
#define STEP_THREADS (KSPLIT * 32)
#define SMAX_THREADS 256

typedef __attribute__((ext_vector_type(16))) __bf16 bf16x16;
typedef __attribute__((ext_vector_type(8)))  float  f32x8;
typedef __attribute__((ext_vector_type(16))) float  f32x16;

// ---------------------------------------------------------------------------
// Utility kernels
// ---------------------------------------------------------------------------
__global__ void cvt_f32_to_bf16(const float* __restrict__ src,
                                __bf16* __restrict__ dst, int n) {
    int i = blockIdx.x * blockDim.x + threadIdx.x;
    if (i < n) dst[i] = (__bf16)src[i];
}

__global__ void zero_bf16(__bf16* __restrict__ dst, int n) {
    int i = blockIdx.x * blockDim.x + threadIdx.x;
    if (i < n) dst[i] = (__bf16)0.0f;
}

// emb[(t*BATCH + b)*EMBED + e] = i2e_w[e*VOCAB + tok(b,t)] + i2e_b[e]
// 128 threads/block, EMBED=512 -> exactly 4 elements per thread.
__global__ void embed_gather(const int* __restrict__ tokens,
                             const float* __restrict__ i2e_w,
                             const float* __restrict__ i2e_b,
                             __bf16* __restrict__ emb) {
    const int bt = blockIdx.x;            // 0 .. B*T-1
    const int b  = bt / TLEN;
    const int t  = bt % TLEN;
    const int tok = tokens[b * TLEN + t];
    __bf16* dst = emb + ((size_t)t * BATCH + b) * EMBED;
    #pragma unroll
    for (int r = 0; r < EMBED / 128; ++r) {
        const int e = threadIdx.x + r * 128;
        float v = i2e_w[(size_t)e * VOCAB + tok] + i2e_b[e];
        dst[e] = (__bf16)v;
    }
}

// ---------------------------------------------------------------------------
// One recurrent step: h_next = [e_t | h_prev] @ Wh^T + bias
// Grid: (HIDDEN/16, BATCH/16), block = 4 waves. Each wave computes the same
// 16x16 C tile over a disjoint K/4 chunk (12 dependent WMMAs instead of 48),
// then partials are reduced through LDS (ds_store_2addr + barrier). This
// shortens the dependent-WMMA chain on the sequential scan path 4x and gives
// 1024 live waves per step.
// ---------------------------------------------------------------------------
__global__ void __launch_bounds__(STEP_THREADS)
rnn_step(const __bf16* __restrict__ e_t,    // (B, EMBED)
         const __bf16* __restrict__ h_prev, // (B, HIDDEN)
         const __bf16* __restrict__ wh,     // (HIDDEN, KTOT)
         const float*  __restrict__ bias,   // (HIDDEN)
         __bf16* __restrict__ h_next) {     // (B, HIDDEN)
    const int nTile = blockIdx.x;               // 0..63
    const int mTile = blockIdx.y;               // 0..3
    const int lane  = threadIdx.x & 31;
    const int wave  = threadIdx.x >> 5;         // 0..3
    const int rowA  = mTile * 16 + (lane & 15); // batch row
    const int rowB  = nTile * 16 + (lane & 15); // output column == Wh row
    const int kHalf = (lane >> 4) * 16;
    const int k0    = wave * KCHUNK;

    const __bf16* aEbase = e_t    + (size_t)rowA * EMBED  + kHalf;
    const __bf16* aHbase = h_prev + (size_t)rowA * HIDDEN + kHalf - EMBED;
    const __bf16* bbase  = wh     + (size_t)rowB * KTOT   + kHalf;

    f32x8 acc = {};
    #pragma unroll 4
    for (int k = k0; k < k0 + KCHUNK; k += 32) {
        const __bf16* ap = (k < EMBED) ? (aEbase + k) : (aHbase + k);
        bf16x16 a = *(const bf16x16*)ap;
        bf16x16 b = *(const bf16x16*)(bbase + k);
        acc = __builtin_amdgcn_wmma_f32_16x16x32_bf16(
                  false, a, false, b, (short)0, acc, false, false);
    }

    // Reduce 4 partial 16x16 tiles through LDS.
    // C/D layout: lanes 0-15 -> M = vgpr, N = lane; lanes 16-31 -> M = 8+vgpr.
    __shared__ float part[KSPLIT][256];
    const int mLoc = (lane >> 4) * 8;
    const int nLoc = lane & 15;
    #pragma unroll
    for (int i = 0; i < 8; ++i)
        part[wave][(mLoc + i) * 16 + nLoc] = acc[i];
    __syncthreads();

    // 256 outputs, 128 threads -> exactly 2 per thread (no runtime trip math).
    #pragma unroll
    for (int r = 0; r < 256 / STEP_THREADS; ++r) {
        const int e = threadIdx.x + r * STEP_THREADS;
        float v = part[0][e] + part[1][e] + part[2][e] + part[3][e];
        const int m = mTile * 16 + (e >> 4);
        const int n = nTile * 16 + (e & 15);
        h_next[(size_t)m * HIDDEN + n] = (__bf16)(v + bias[n]);
    }
}

// ---------------------------------------------------------------------------
// logits = [e_last | h_last] @ i2o_w^T + i2o_b
// Grid: OUTD/16 blocks, block = 1 wave computing a 64x16 slab (4 C tiles).
// Streams f32 weights from HBM (memory-bound: 196 MB @ 23.3 TB/s), converts
// to bf16 in registers for WMMA.
// ---------------------------------------------------------------------------
__global__ void __launch_bounds__(32)
proj_logits(const __bf16* __restrict__ e_last, // (B, EMBED)
            const __bf16* __restrict__ h_last, // (B, HIDDEN)
            const float*  __restrict__ w,      // (OUTD, KTOT)
            const float*  __restrict__ bias,   // (OUTD)
            float* __restrict__ logits) {      // (B, OUTD)
    const int nTile = blockIdx.x;          // 0..1999
    const int lane  = threadIdx.x;
    const int rowB  = nTile * 16 + (lane & 15);
    const int kHalf = (lane >> 4) * 16;
    const int rA    = lane & 15;

    const float* wp = w + (size_t)rowB * KTOT + kHalf;

    f32x8 acc0 = {}, acc1 = {}, acc2 = {}, acc3 = {};

    #pragma unroll 2
    for (int k = 0; k < EMBED; k += 32) {
        __builtin_prefetch(wp + k + 128, 0, 0);
        f32x16 wf = *(const f32x16*)(wp + k);
        bf16x16 b;
        #pragma unroll
        for (int i = 0; i < 16; ++i) b[i] = (__bf16)wf[i];
        const __bf16* ar = e_last + kHalf + k;
        bf16x16 a0 = *(const bf16x16*)(ar + (size_t)(rA +  0) * EMBED);
        bf16x16 a1 = *(const bf16x16*)(ar + (size_t)(rA + 16) * EMBED);
        bf16x16 a2 = *(const bf16x16*)(ar + (size_t)(rA + 32) * EMBED);
        bf16x16 a3 = *(const bf16x16*)(ar + (size_t)(rA + 48) * EMBED);
        acc0 = __builtin_amdgcn_wmma_f32_16x16x32_bf16(false, a0, false, b, (short)0, acc0, false, false);
        acc1 = __builtin_amdgcn_wmma_f32_16x16x32_bf16(false, a1, false, b, (short)0, acc1, false, false);
        acc2 = __builtin_amdgcn_wmma_f32_16x16x32_bf16(false, a2, false, b, (short)0, acc2, false, false);
        acc3 = __builtin_amdgcn_wmma_f32_16x16x32_bf16(false, a3, false, b, (short)0, acc3, false, false);
    }
    #pragma unroll 2
    for (int k = 0; k < HIDDEN; k += 32) {
        __builtin_prefetch(wp + EMBED + k + 128, 0, 0);
        f32x16 wf = *(const f32x16*)(wp + EMBED + k);
        bf16x16 b;
        #pragma unroll
        for (int i = 0; i < 16; ++i) b[i] = (__bf16)wf[i];
        const __bf16* ar = h_last + kHalf + k;
        bf16x16 a0 = *(const bf16x16*)(ar + (size_t)(rA +  0) * HIDDEN);
        bf16x16 a1 = *(const bf16x16*)(ar + (size_t)(rA + 16) * HIDDEN);
        bf16x16 a2 = *(const bf16x16*)(ar + (size_t)(rA + 32) * HIDDEN);
        bf16x16 a3 = *(const bf16x16*)(ar + (size_t)(rA + 48) * HIDDEN);
        acc0 = __builtin_amdgcn_wmma_f32_16x16x32_bf16(false, a0, false, b, (short)0, acc0, false, false);
        acc1 = __builtin_amdgcn_wmma_f32_16x16x32_bf16(false, a1, false, b, (short)0, acc1, false, false);
        acc2 = __builtin_amdgcn_wmma_f32_16x16x32_bf16(false, a2, false, b, (short)0, acc2, false, false);
        acc3 = __builtin_amdgcn_wmma_f32_16x16x32_bf16(false, a3, false, b, (short)0, acc3, false, false);
    }

    const int nCol  = nTile * 16 + (lane & 15);
    const int mOff  = (lane >> 4) * 8;
    const float bv  = bias[nCol];
    #pragma unroll
    for (int i = 0; i < 8; ++i) {
        logits[(size_t)( 0 + mOff + i) * OUTD + nCol] = acc0[i] + bv;
        logits[(size_t)(16 + mOff + i) * OUTD + nCol] = acc1[i] + bv;
        logits[(size_t)(32 + mOff + i) * OUTD + nCol] = acc2[i] + bv;
        logits[(size_t)(48 + mOff + i) * OUTD + nCol] = acc3[i] + bv;
    }
}

// ---------------------------------------------------------------------------
// Row softmax over 32000 columns. One 256-thread block per batch row.
// Compile-time stride (256) everywhere -> no runtime trip-count math.
// ---------------------------------------------------------------------------
__global__ void __launch_bounds__(SMAX_THREADS)
softmax_rows(const float* __restrict__ logits,
             float* __restrict__ out) {
    const int row = blockIdx.x;
    const float* src = logits + (size_t)row * OUTD;
    float* dst = out + (size_t)row * OUTD;

    __shared__ float redMax[8];
    __shared__ float redSum[8];
    const int wid = threadIdx.x >> 5;
    const int ln  = threadIdx.x & 31;

    float m = -3.4e38f;
    for (int i = threadIdx.x; i < OUTD; i += SMAX_THREADS)
        m = fmaxf(m, src[i]);
    #pragma unroll
    for (int off = 16; off > 0; off >>= 1)
        m = fmaxf(m, __shfl_xor(m, off, 32));
    if (ln == 0) redMax[wid] = m;
    __syncthreads();
    if (threadIdx.x == 0) {
        float mm = redMax[0];
        #pragma unroll
        for (int i = 1; i < 8; ++i) mm = fmaxf(mm, redMax[i]);
        redMax[0] = mm;
    }
    __syncthreads();
    m = redMax[0];

    float s = 0.0f;
    for (int i = threadIdx.x; i < OUTD; i += SMAX_THREADS)
        s += __expf(src[i] - m);
    #pragma unroll
    for (int off = 16; off > 0; off >>= 1)
        s += __shfl_xor(s, off, 32);
    if (ln == 0) redSum[wid] = s;
    __syncthreads();
    if (threadIdx.x == 0) {
        float ss = redSum[0];
        #pragma unroll
        for (int i = 1; i < 8; ++i) ss += redSum[i];
        redSum[0] = ss;
    }
    __syncthreads();
    const float inv = 1.0f / redSum[0];

    for (int i = threadIdx.x; i < OUTD; i += SMAX_THREADS)
        dst[i] = __expf(src[i] - m) * inv;
}

// ---------------------------------------------------------------------------
// Host orchestration: all launches on `stream`, graph-capture safe.
// ---------------------------------------------------------------------------
extern "C" void kernel_launch(void* const* d_in, const int* in_sizes, int n_in,
                              void* d_out, int out_size, void* d_ws, size_t ws_size,
                              hipStream_t stream) {
    const int*   tokens = (const int*)  d_in[0];
    const float* i2e_w  = (const float*)d_in[1];
    const float* i2e_b  = (const float*)d_in[2];
    const float* i2o_w  = (const float*)d_in[3];
    const float* i2o_b  = (const float*)d_in[4];
    const float* i2h_w  = (const float*)d_in[5];
    const float* i2h_b  = (const float*)d_in[6];

    char* ws = (char*)d_ws;
    size_t off = 0;
    __bf16* emb = (__bf16*)(ws + off);  off += (size_t)TLEN * BATCH * EMBED * 2;  // 32 MB
    __bf16* whb = (__bf16*)(ws + off);  off += (size_t)HIDDEN * KTOT * 2;         //  3 MB
    __bf16* h0  = (__bf16*)(ws + off);  off += (size_t)BATCH * HIDDEN * 2;
    __bf16* h1  = (__bf16*)(ws + off);  off += (size_t)BATCH * HIDDEN * 2;
    float* logits = (float*)(ws + off);                                           //  8 MB

    // Recurrent weights -> bf16 (L2-resident thereafter).
    const int nW = HIDDEN * KTOT;
    cvt_f32_to_bf16<<<(nW + 255) / 256, 256, 0, stream>>>(i2h_w, whb, nW);

    // Embedding gather (table is L2-resident).
    embed_gather<<<BATCH * TLEN, 128, 0, stream>>>(tokens, i2e_w, i2e_b, emb);

    // h0 = 0 (must re-zero every call; ws is not re-poisoned).
    const int nH = BATCH * HIDDEN;
    zero_bf16<<<(nH + 255) / 256, 256, 0, stream>>>(h0, nH);

    // Sequential scan: 511 dependent step GEMMs, ping-ponged hidden state.
    __bf16* hp = h0;
    __bf16* hn = h1;
    for (int t = 0; t < TLEN - 1; ++t) {
        const __bf16* e_t = emb + (size_t)t * BATCH * EMBED;
        rnn_step<<<dim3(HIDDEN / 16, BATCH / 16), STEP_THREADS, 0, stream>>>(
            e_t, hp, whb, i2h_b, hn);
        __bf16* tmp = hp; hp = hn; hn = tmp;
    }

    // Output projection (streams 196 MB of f32 weights once from HBM).
    const __bf16* e_last = emb + (size_t)(TLEN - 1) * BATCH * EMBED;
    proj_logits<<<OUTD / 16, 32, 0, stream>>>(e_last, hp, i2o_w, i2o_b, logits);

    // Softmax per batch row.
    softmax_rows<<<BATCH, SMAX_THREADS, 0, stream>>>(logits, (float*)d_out);
}
